// BatchMultiHeadGraphAttention_24773371363431
// MI455X (gfx1250) — matrix operand order repaired and checked by reading, not compiled, not run on hardware
//
#include <hip/hip_runtime.h>

// ---------------------------------------------------------------------------
// BatchMultiHeadGraphAttention for MI455X (gfx1250, wave32, WMMA).
// BS=4, N=2048, F_IN=128, F_OUT=64, NH=8, NTYPE=3
//
// proj_kernel: h_prime = h @ w via v_wmma_f32_16x16x32_bf16; tanh + per-type
//   src/dst score dots. h_prime stored bf16 transposed [b][h][o][n].
// attn_kernel: masked softmax over rank-1 logits lrelu(src_i+dst_j), fused
//   with P @ h_prime via WMMA. Key optimizations:
//   - exp computed exactly ONCE per (i,j): unnormalized P~ accumulated by
//     WMMA, row-sum s_i accumulated concurrently (ds_add_f32), normalization
//     applied as a per-row scalar AFTER accumulation.
//   - row max via leaky-ReLU monotonicity: m_i = lrelu(src_i + max_adj dst_j)
//     (no exp in pass 1).
//   - adjacency rows staged LDS via gfx1250 async copy
//     (global_load_async_to_lds_b128 + s_wait_asynccnt).
// ---------------------------------------------------------------------------

static constexpr int kBS    = 4;
static constexpr int kN     = 2048;
static constexpr int kFIN   = 128;
static constexpr int kFOUT  = 64;
static constexpr int kNH    = 8;
static constexpr int kNTYPE = 3;

typedef __attribute__((ext_vector_type(16))) __bf16 v16bf;
typedef __attribute__((ext_vector_type(8)))  float  v8f;

// 16-bit A-matrix 16x32 element->K mapping (CDNA5 ISA 7.12.2):
// lanes 0-15: VGPR0..3 K=0..7 (2 packed), VGPR4..7 K=16..23
// lanes16-31: VGPR0..3 K=8..15,           VGPR4..7 K=24..31
__device__ __forceinline__ int a_k_off(int e, int half) {
  int v = e >> 1, p = e & 1;
  int base = ((v < 4) ? 0 : 16) + half * 8;
  return base + ((v & 3) << 1) + p;
}
// 16-bit B 32x16 as loaded here (A alike, contiguous): lanes 0-15 K=0..15, lanes 16-31 K=16..31 -> K=half*16+e

// gfx1250 async copy: 16B global -> LDS, tracked by ASYNCcnt.
__device__ __forceinline__ void async_ld_b128(void* lds_ptr, const void* gptr) {
  // low 32 bits of the generic (LDS-aperture) address == LDS byte offset
  unsigned int loff = (unsigned int)(unsigned long long)lds_ptr;
  asm volatile("global_load_async_to_lds_b128 %0, %1, off"
               :: "v"(loff), "v"(gptr) : "memory");
}
__device__ __forceinline__ void wait_async0() {
  asm volatile("s_wait_asynccnt 0" ::: "memory");
}

// ---------------------------------------------------------------------------
// Kernel 1: projection + tanh + per-type scores.
// grid = BS*NH*(N/64), block = 256 (8 waves). Each block: 64 nodes x 64 out.
// ---------------------------------------------------------------------------
__global__ __launch_bounds__(256) void proj_kernel(
    const float* __restrict__ h,        // [BS][N][F_IN]
    const float* __restrict__ w,        // [NH][F_IN][F_OUT]
    const float* __restrict__ v_types,  // [BS][N][NTYPE] one-hot
    const float* __restrict__ a_src,    // [NH][F_OUT][NTYPE]
    const float* __restrict__ a_dst,    // [NH][F_OUT][NTYPE]
    unsigned short* __restrict__ hp_raw, // bf16 [BS][NH][F_OUT][N] (transposed)
    float* __restrict__ src_s,          // [BS][NH][N]
    float* __restrict__ dst_s)          // [BS][NH][N]
{
  __bf16* hp_t = (__bf16*)hp_raw;
  __shared__ __bf16 hs[64][kFIN];      // 16 KB
  __shared__ __bf16 wts[kFOUT][kFIN];  // 16 KB (w transposed)
  __shared__ float  tns[64][kFOUT];    // 16 KB tanh tile

  const int blk   = blockIdx.x;
  const int ntile = blk % (kN / 64);
  const int head  = (blk / (kN / 64)) % kNH;
  const int b     = blk / ((kN / 64) * kNH);
  const int n0    = ntile * 64;
  const int tid   = threadIdx.x;

  for (int i = tid; i < 64 * kFIN; i += 256) {
    int r = i / kFIN, k = i % kFIN;
    hs[r][k] = (__bf16)h[((size_t)b * kN + n0 + r) * kFIN + k];
  }
  for (int i = tid; i < kFIN * kFOUT; i += 256) {
    int k = i / kFOUT, o = i % kFOUT;
    wts[o][k] = (__bf16)w[((size_t)head * kFIN + k) * kFOUT + o];
  }
  __syncthreads();

  const int wave = tid >> 5, lane = tid & 31;
  const int half = lane >> 4, lr = lane & 15;
  const int rowg  = wave >> 1;
  const int colg0 = (wave & 1) * 2;

  v8f acc0 = {}; v8f acc1 = {};
  for (int k0 = 0; k0 < kFIN; k0 += 32) {
    v16bf a, b0, b1;
#pragma unroll
    for (int e = 0; e < 16; ++e) {
      a[e]  = hs[rowg * 16 + lr][k0 + half * 16 + e];
      int kb = k0 + half * 16 + e;
      b0[e] = wts[colg0 * 16 + lr][kb];
      b1[e] = wts[(colg0 + 1) * 16 + lr][kb];
    }
    acc0 = __builtin_amdgcn_wmma_f32_16x16x32_bf16(false, a, false, b0,
                                                   (short)0, acc0, false, false);
    acc1 = __builtin_amdgcn_wmma_f32_16x16x32_bf16(false, a, false, b1,
                                                   (short)0, acc1, false, false);
  }

  const size_t base = ((size_t)b * kNH + head) * kFOUT;
#pragma unroll
  for (int r = 0; r < 8; ++r) {
    int m  = rowg * 16 + r + half * 8;
    int o0 = colg0 * 16 + lr;
    int o1 = (colg0 + 1) * 16 + lr;
    float x0 = acc0[r], x1 = acc1[r];
    tns[m][o0] = tanhf(x0);
    tns[m][o1] = tanhf(x1);
    hp_t[(base + o0) * kN + n0 + m] = (__bf16)x0;
    hp_t[(base + o1) * kN + n0 + m] = (__bf16)x1;
  }
  __syncthreads();

  if (tid < 64) {
    int node = n0 + tid;
    const float* vt = &v_types[((size_t)b * kN + node) * kNTYPE];
    int t = (vt[1] > 0.5f) ? 1 : ((vt[2] > 0.5f) ? 2 : 0);
    const float* as = &a_src[(size_t)head * kFOUT * kNTYPE + t];
    const float* ad = &a_dst[(size_t)head * kFOUT * kNTYPE + t];
    float ssum = 0.f, dsum = 0.f;
#pragma unroll 4
    for (int o = 0; o < kFOUT; ++o) {
      float tv = tns[tid][o];
      ssum += tv * as[o * kNTYPE];
      dsum += tv * ad[o * kNTYPE];
    }
    size_t idx = ((size_t)b * kNH + head) * kN + node;
    src_s[idx] = ssum;
    dst_s[idx] = dsum;
  }
}

// ---------------------------------------------------------------------------
// Kernel 2: masked softmax + WMMA aggregation, 1 exp per (i,j).
// grid = BS*NH*(N/16), block = 256 (8 waves). Each block: 16 rows x 64 out.
// ---------------------------------------------------------------------------
__global__ __launch_bounds__(256) void attn_kernel(
    const unsigned char* __restrict__ adj,     // [BS][N][N] bool (1 byte)
    const unsigned short* __restrict__ hp_raw, // bf16 [BS][NH][F_OUT][N]
    const float* __restrict__ src_s,
    const float* __restrict__ dst_s,
    const float* __restrict__ bias,            // [F_OUT]
    float* __restrict__ out)                   // [BS][NH][N][F_OUT] f32
{
  const __bf16* hp_t = (const __bf16*)hp_raw;
  __shared__ float dsh[kN];                            // 8 KB dst scores
  __shared__ __align__(16) unsigned char ash[16][kN];  // 32 KB adj rows
  __shared__ __align__(32) __bf16 pt[16][256];         // 8 KB P~ superstep tile
  __shared__ float msh[16], ssh[16], srcsh[16];
  __shared__ float osh[16][kFOUT];                     // 4 KB half-reduce

  const int blk   = blockIdx.x;
  const int itile = blk % (kN / 16);
  const int head  = (blk / (kN / 16)) % kNH;
  const int b     = blk / ((kN / 16) * kNH);
  const int i0    = itile * 16;
  const int tid   = threadIdx.x;
  const size_t bh = (size_t)b * kNH + head;

  // ---- stage 0: async-copy adj rows to LDS; load dst/src scores ----------
  const uint4* a4 = (const uint4*)&adj[((size_t)b * kN + i0) * kN];
  uint4* d4 = (uint4*)&ash[0][0];
  for (int i = tid; i < 16 * kN / 16; i += 256)
    async_ld_b128((void*)(d4 + i), (const void*)(a4 + i));
  for (int j = tid; j < kN; j += 256) dsh[j] = dst_s[bh * kN + j];
  if (tid < 16) { srcsh[tid] = src_s[bh * kN + i0 + tid]; ssh[tid] = 0.f; }
  wait_async0();
  __syncthreads();

  const int wave = tid >> 5, lane = tid & 31;

  // ---- pass 1: masked max of dst_j (lrelu is monotone => no exp needed) ---
  for (int rr = 0; rr < 2; ++rr) {
    int row = wave * 2 + rr;
    float mx = -1e30f;
    for (int j = lane; j < kN; j += 32)
      if (ash[row][j]) mx = fmaxf(mx, dsh[j]);
#pragma unroll
    for (int off = 16; off > 0; off >>= 1) mx = fmaxf(mx, __shfl_xor(mx, off, 32));
    if (lane == 0) {
      float l = srcsh[row] + mx;
      msh[row] = (l >= 0.f) ? l : 0.2f * l;   // garbage for empty rows: unused
    }
  }
  __syncthreads();

  // ---- pass 2: supersteps of 256 j. Build P~ once (8 waves x 32 j), then
  //      4 col-group x 2 j-half waves run WMMA over the shared tile. --------
  const int half = lane >> 4, lr = lane & 15;
  const int colg = wave & 3;
  const int jh   = wave >> 2;
  const int brow = lane & 15;            // build: row handled by this lane
  const int bj   = (lane >> 4) * 16;     // build: j sub-half within 32-chunk
  const float srcb = srcsh[brow];

  v8f acc = {};
  const __bf16* hpb = &hp_t[(bh * kFOUT + colg * 16 + lr) * kN];

  for (int js = 0; js < kN; js += 256) {
    // build this wave's 16x32 chunk of unnormalized probabilities
    {
      const float mb = msh[brow];
      const int loffb = wave * 32;
      float lsum = 0.f;
      v16bf tmp;
#pragma unroll
      for (int t = 0; t < 16; ++t) {
        int jl = loffb + bj + t;
        int j  = js + jl;
        float p = 0.f;
        if (ash[brow][j]) {
          float l = srcb + dsh[j];
          l = (l >= 0.f) ? l : 0.2f * l;
          p = __expf(l - mb);
        }
        lsum += p;
        tmp[t] = (__bf16)p;
      }
      *(v16bf*)&pt[brow][loffb + bj] = tmp;
      lsum += __shfl_xor(lsum, 16, 32);
      if (lane < 16) atomicAdd(&ssh[lane], lsum);
    }
    __syncthreads();

    // WMMA: this wave covers 4 chunks (128 j) of the shared P~ tile
#pragma unroll
    for (int c = 0; c < 4; ++c) {
      int loff = jh * 128 + c * 32;
      int j0g  = js + loff;
      v16bf pa, hb;
#pragma unroll
      for (int e = 0; e < 16; ++e) {
        pa[e] = pt[lr][loff + half * 16 + e];
        hb[e] = hpb[j0g + half * 16 + e];
      }
      acc = __builtin_amdgcn_wmma_f32_16x16x32_bf16(false, pa, false, hb,
                                                    (short)0, acc, false, false);
    }
    __syncthreads();
  }

  // ---- reduce j-halves, normalize by row sum, add bias, store f32 --------
  if (jh == 1) {
#pragma unroll
    for (int r = 0; r < 8; ++r) osh[r + half * 8][colg * 16 + lr] = acc[r];
  }
  __syncthreads();
  if (jh == 0) {
#pragma unroll
    for (int r = 0; r < 8; ++r) {
      int m = r + half * 8;
      int o = colg * 16 + lr;
      float s  = ssh[m];
      float sc = (s > 0.f) ? 1.0f / s : 0.f;   // empty row -> bias only
      out[(bh * kN + i0 + m) * kFOUT + o] = (acc[r] + osh[m][o]) * sc + bias[o];
    }
  }
}

// ---------------------------------------------------------------------------
extern "C" void kernel_launch(void* const* d_in, const int* in_sizes, int n_in,
                              void* d_out, int out_size, void* d_ws, size_t ws_size,
                              hipStream_t stream) {
  const float*         h       = (const float*)d_in[0];
  const unsigned char* adj     = (const unsigned char*)d_in[1]; // bool, 1 byte
  const float*         v_types = (const float*)d_in[2];
  const float*         w       = (const float*)d_in[3];
  const float*         a_src   = (const float*)d_in[4];
  const float*         a_dst   = (const float*)d_in[5];
  const float*         bias    = (const float*)d_in[6];
  float*               out     = (float*)d_out;

  char* ws = (char*)d_ws;
  const size_t hp_bytes = (size_t)kBS * kNH * kFOUT * kN * sizeof(unsigned short);
  unsigned short* hp_t  = (unsigned short*)ws;
  float* src_s = (float*)(ws + hp_bytes);
  float* dst_s = src_s + (size_t)kBS * kNH * kN;

  dim3 block(256);
  proj_kernel<<<dim3(kBS * kNH * (kN / 64)), block, 0, stream>>>(
      h, w, v_types, a_src, a_dst, hp_t, src_s, dst_s);
  attn_kernel<<<dim3(kBS * kNH * (kN / 16)), block, 0, stream>>>(
      adj, hp_t, src_s, dst_s, bias, out);
}
